// HybridQLSTM_65481071408714
// MI455X (gfx1250) — matrix operand order repaired and checked by reading, not compiled, not run on hardware
//
#include <hip/hip_runtime.h>
#include <hip/hip_bf16.h>
#include <math.h>
#include <stdint.h>

// ---------------- problem constants ----------------
#define T_STEPS 2048
#define BATCH   64
#define DIN     256
#define DH      256
#define NPAD    1040              // 4*256 gates + 8 quantum + 8 zero pad
#define NT      65                // NPAD / 16
#define MT_ALL  (T_STEPS * BATCH / 16)   // 8192 m-tiles for input projection
#define Q_STRENGTH 0.5f
#define SLAB_HALVES (4 * NT * 256)       // 66,560 f16 per timestep slab (133,120 B)

typedef __attribute__((ext_vector_type(16))) _Float16 v16h;
typedef __attribute__((ext_vector_type(8)))  _Float16 v8h;
typedef __attribute__((ext_vector_type(8)))  float    v8f;
typedef unsigned int u32x4 __attribute__((ext_vector_type(4)));
typedef int          i32x4 __attribute__((ext_vector_type(4)));
typedef int          i32x8 __attribute__((ext_vector_type(8)));

union ABu { v16h v; v8h h[2]; };

// ---- WMMA fragment helpers (layouts per CDNA5 ISA 7.12.2, wave32) ----
// A 16x32 f16: lanes 0-15 hold row M=lane, elems 0..7 = K h*8+0..7, elems 8..15 = K 16+h*8+0..7
__device__ __forceinline__ v16h load_a_frag(const _Float16* base, int ld, int lane) {
  const int hh = lane >> 4, m = lane & 15;
  const _Float16* p = base + m * ld + hh * 8;
  ABu ab;
  ab.h[0] = *(const v8h*)(p);
  ab.h[1] = *(const v8h*)(p + 16);
  return ab.v;
}

// B 32x16 f16 fragment, pre-packed contiguously per lane (16 f16 per lane)
__device__ __forceinline__ v16h load_b_frag(const _Float16* tile, int lane) {
  const v8h* p = (const v8h*)(tile + lane * 16);
  ABu ab; ab.h[0] = p[0]; ab.h[1] = p[1];
  return ab.v;
}

__device__ __forceinline__ v8f wmma16x16x32(v16h a, v16h b, v8f c) {
  return __builtin_amdgcn_wmma_f32_16x16x32_f16(false, a, false, b, (short)0, c, false, false);
}

__device__ __forceinline__ float sigmoidf_(float x) { return 1.0f / (1.0f + __expf(-x)); }

// ---- Tensor Data Mover: 1-D slab copy global -> LDS (D# per ISA ch.8) ----
// n8 = number of 8-byte elements (data_size code 3). Issued once per wave; TENSORcnt-tracked.
__device__ __forceinline__ void tdm_load_slab(const void* gsrc, void* lds_dst, int n8) {
  uint64_t ga = (uint64_t)(uintptr_t)gsrc;
  uint32_t lo = (uint32_t)(uintptr_t)lds_dst;     // flat LDS aperture: addr[31:0] = LDS offset
  u32x4 g0;
  g0[0] = 1u;                                     // count=1, user descriptor
  g0[1] = lo;                                     // lds_addr (bytes)
  g0[2] = (uint32_t)ga;                           // global_addr[31:0]
  g0[3] = ((uint32_t)(ga >> 32) & 0x01FFFFFFu) | (2u << 30);  // addr[56:32], type=2
  i32x8 g1;
  g1[0] = (int)(3u << 16);                        // data_size=3 (8B); no multicast/pad/iterate
  g1[1] = (int)((n8 & 0xFFFF) << 16);             // tensor_dim0[15:0]  (bits 63:48)
  g1[2] = (int)(((n8 >> 16) & 0xFFFF) | (1 << 16)); // tensor_dim0[31:16] | tensor_dim1=1
  g1[3] = (int)((n8 & 0xFFFF) << 16);             // tile_dim0 (bits 127:112)
  g1[4] = 1;                                      // tile_dim1=1, tile_dim2=0
  g1[5] = n8;                                     // tensor_dim0_stride[31:0]
  g1[6] = 0;
  g1[7] = 0;
  i32x4 z4 = {0, 0, 0, 0};
#if defined(__clang_major__) && (__clang_major__ >= 23)
  i32x8 z8 = {0, 0, 0, 0, 0, 0, 0, 0};
  __builtin_amdgcn_tensor_load_to_lds(g0, g1, z4, z4, z8, 0);
#else
  __builtin_amdgcn_tensor_load_to_lds(g0, g1, z4, z4, 0);
#endif
}

// ---------------- kernel 1: fp32 -> f16 input conversion ----------------
__global__ void k_convert_x(const float* __restrict__ x, _Float16* __restrict__ xh, int n) {
  for (int i = blockIdx.x * blockDim.x + threadIdx.x; i < n; i += gridDim.x * blockDim.x)
    xh[i] = (_Float16)x[i];
}

// ---------------- kernel 2: pack weights into B-fragment layout + bias ----------------
// wpack layout: [part(2: x-rows 0..255 / h-rows 256..511)][kt(8)][nt(65)][lane(32)][e(16)] f16
__global__ void k_pack(const float* __restrict__ Wf, const float* __restrict__ Wi,
                       const float* __restrict__ Wg, const float* __restrict__ Wo,
                       const float* __restrict__ Wq,
                       const float* __restrict__ bf, const float* __restrict__ bi,
                       const float* __restrict__ bg, const float* __restrict__ bo,
                       const float* __restrict__ bq,
                       _Float16* __restrict__ wpack, float* __restrict__ bias) {
  const int total = 2 * 8 * NT * 512;
  for (int idx = blockIdx.x * blockDim.x + threadIdx.x; idx < total;
       idx += gridDim.x * blockDim.x) {
    int e    = idx & 15;
    int lane = (idx >> 4) & 31;
    int tile = idx >> 9;                 // part*8*65 + kt*65 + nt
    int nt   = tile % NT;
    int ktp  = tile / NT;
    int kt   = ktp & 7;
    int part = ktp >> 3;
    int k = part * 256 + kt * 32 + ((lane >> 4) * 8) + (e & 7) + ((e >> 3) * 16);   // 0..511 combined row
    int n = nt * 16 + (lane & 15);                           // 0..1039 output col
    float w = 0.0f;
    if (n < 1024) {
      const float* W = (n < 256) ? Wf : (n < 512) ? Wi : (n < 768) ? Wg : Wo;
      w = W[k * 256 + (n & 255)];
    } else if (n < 1032) {
      w = Wq[k * 8 + (n - 1024)];
    }
    wpack[idx] = (_Float16)w;
  }
  for (int n = blockIdx.x * blockDim.x + threadIdx.x; n < NPAD;
       n += gridDim.x * blockDim.x) {
    float b = 0.0f;
    if      (n < 256)  b = bf[n];
    else if (n < 512)  b = bi[n - 256];
    else if (n < 768)  b = bg[n - 512];
    else if (n < 1024) b = bo[n - 768];
    else if (n < 1032) b = bq[n - 1024];
    bias[n] = b;
  }
}

// ---------------- kernel 3: input projection Xproj = x @ Wx + bias ----------------
// Output stored in C-fragment order: [tile = mt*65+nt][lane*8 + r] f16
__global__ void k_input_proj(const _Float16* __restrict__ xh,
                             const _Float16* __restrict__ wpack,
                             const float* __restrict__ bias,
                             _Float16* __restrict__ xproj) {
  const int wave = blockIdx.x * (blockDim.x >> 5) + (threadIdx.x >> 5);
  const int lane = threadIdx.x & 31;
  const int mt = wave / NT, nt = wave % NT;
  if (mt >= MT_ALL) return;
  const int row0 = mt * 16, n0 = nt * 16;

  float bv = bias[n0 + (lane & 15)];
  v8f acc = {bv, bv, bv, bv, bv, bv, bv, bv};

  const _Float16* abase = xh + (size_t)row0 * DIN;
#pragma unroll
  for (int kt = 0; kt < 8; ++kt) {
    v16h a = load_a_frag(abase + kt * 32, DIN, lane);
    v16h b = load_b_frag(wpack + ((size_t)(kt * NT + nt)) * 512, lane);
    acc = wmma16x16x32(a, b, acc);
  }
  v8h o;
#pragma unroll
  for (int r = 0; r < 8; ++r) o[r] = (_Float16)acc[r];
  *(v8h*)(xproj + (size_t)wave * 256 + lane * 8) = o;
}

// ---------------- kernel 4: persistent recurrent scan (1 WGP, 32 waves) ----------------
// wave w (0..31): nt = w>>1 (h-column tile 0..15), m-tiles {mb, mb+16} with mb=(w&1)*32.
// Each wave holds 2x4 gate accumulators + 2 c-state tiles in VGPRs.
// Waves 0..3 additionally compute the quantum tile (nt=64) for m-tiles 0,32,16,48.
// Per-step Xproj slab (133 KB) streamed into a ping-pong LDS buffer with the TDM,
// overlapped with the recurrent WMMA GEMM; drained with s_wait_tensorcnt.
#define HLD 264   // LDS row stride in halves (bank-conflict-free for b128 column reads)

__global__ __launch_bounds__(1024, 1)
void k_recurrent(const _Float16* __restrict__ wh,     // packed part-1 tiles [kt][nt][512]
                 const _Float16* __restrict__ xproj,  // fragment-major, per t: 4*65 tiles
                 float* __restrict__ out,             // stacked [T,B,DH]
                 float* __restrict__ out_hx, float* __restrict__ out_cx) {
  __shared__ __align__(16) _Float16 xbuf[2][SLAB_HALVES];   // 2 x 133,120 B
  __shared__ __align__(16) _Float16 h_lds[64 * HLD];        // 33,792 B
  __shared__ float q_lds[64];

  const int tid = threadIdx.x, lane = tid & 31, w = tid >> 5;
  const int hh = lane >> 4, ln = lane & 15;
  const int nt = w >> 1;
  const int mb = (w & 1) * 32;
  const int m0[2] = {mb, mb + 16};
  const bool has_q = (w < 4);
  const int q_ap = w >> 1;                 // which of this wave's m-tiles carries q

  for (int i = tid; i < 64 * HLD; i += 1024) h_lds[i] = (_Float16)0.0f;

  v8f acc[2][4];
  v8f cst[2];
#pragma unroll
  for (int p = 0; p < 2; ++p)
#pragma unroll
    for (int r = 0; r < 8; ++r) cst[p][r] = 0.0f;

  // preload slab 0 via TDM, then make it visible to all waves
  if (w == 0) {
    tdm_load_slab(xproj, &xbuf[0][0], SLAB_HALVES / 4);
    __builtin_amdgcn_s_wait_tensorcnt(0);
  }
  __syncthreads();

  for (int t = 0; t < T_STEPS; ++t) {
    // kick off async copy of next step's slab into the other LDS buffer
    if (w == 0 && (t + 1) < T_STEPS) {
      tdm_load_slab(xproj + (size_t)(t + 1) * SLAB_HALVES,
                    &xbuf[(t + 1) & 1][0], SLAB_HALVES / 4);
    }

    const _Float16* xp = &xbuf[t & 1][0];

    // accumulator init from x-projection (+bias), fragment layout, from LDS
#pragma unroll
    for (int p = 0; p < 2; ++p) {
      const int mtl = m0[p] >> 4;
#pragma unroll
      for (int g = 0; g < 4; ++g) {
        const int ntg = g * 16 + nt;
        v8h v = *(const v8h*)(xp + ((size_t)(mtl * NT + ntg)) * 256 + lane * 8);
#pragma unroll
        for (int r = 0; r < 8; ++r) acc[p][g][r] = (float)v[r];
      }
    }
    v8f qacc = {};
    if (has_q) {
      const int mtl = m0[q_ap] >> 4;
      v8h v = *(const v8h*)(xp + ((size_t)(mtl * NT + 64)) * 256 + lane * 8);
#pragma unroll
      for (int r = 0; r < 8; ++r) qacc[r] = (float)v[r];
    }

    // ---- h(t-1) @ Wh via WMMA, K=256 ----
#pragma unroll
    for (int kt = 0; kt < 8; ++kt) {
      v16h a0 = load_a_frag(&h_lds[m0[0] * HLD + kt * 32], HLD, lane);
      v16h a1 = load_a_frag(&h_lds[m0[1] * HLD + kt * 32], HLD, lane);
#pragma unroll
      for (int g = 0; g < 4; ++g) {
        v16h b = load_b_frag(wh + ((size_t)(kt * NT + g * 16 + nt)) * 512, lane);
        acc[0][g] = wmma16x16x32(a0, b, acc[0][g]);
        acc[1][g] = wmma16x16x32(a1, b, acc[1][g]);
      }
      if (has_q) {
        v16h bq = load_b_frag(wh + ((size_t)(kt * NT + 64)) * 512, lane);
        qacc = wmma16x16x32(q_ap ? a1 : a0, bq, qacc);
      }
    }

    // ---- quantum gate: sigmoid(sum_n tanh(q_preact)) per row ----
    if (has_q) {
      float s[8];
#pragma unroll
      for (int r = 0; r < 8; ++r) s[r] = tanhf(qacc[r]);   // padded cols give tanh(0)=0
#pragma unroll
      for (int msk = 1; msk < 16; msk <<= 1) {
#pragma unroll
        for (int r = 0; r < 8; ++r) s[r] += __shfl_xor(s[r], msk, 32);
      }
      if (ln == 0) {
        const int rb = m0[q_ap] + 8 * hh;
#pragma unroll
        for (int r = 0; r < 8; ++r) q_lds[rb + r] = sigmoidf_(s[r]);
      }
    }
    __syncthreads();   // q_lds ready; all h_lds reads of this step done

    // ---- activations + state update ----
    const int col = nt * 16 + ln;
#pragma unroll
    for (int p = 0; p < 2; ++p) {
#pragma unroll
      for (int r = 0; r < 8; ++r) {
        const int row = m0[p] + 8 * hh + r;
        float fc = sigmoidf_(acc[p][0][r]);
        float ig = sigmoidf_(acc[p][1][r]);
        float gg = tanhf(acc[p][2][r]);
        float og = sigmoidf_(acc[p][3][r]);
        float f  = (1.0f - Q_STRENGTH) * fc + Q_STRENGTH * q_lds[row];
        float c  = f * cst[p][r] + ig * gg;
        cst[p][r] = c;
        float hv = og * tanhf(c);
        h_lds[row * HLD + col] = (_Float16)hv;
        out[((size_t)t * BATCH + row) * DH + col] = hv;
        if (t == T_STEPS - 1) {
          out_hx[row * DH + col] = hv;
          out_cx[row * DH + col] = c;
        }
      }
    }

    // drain next-slab TDM before anyone flips to the other buffer
    if (w == 0 && (t + 1) < T_STEPS) __builtin_amdgcn_s_wait_tensorcnt(0);
    __syncthreads();   // new h + next slab visible before next step
  }
}

// ---------------- host launcher ----------------
// Workspace requirement: ~341 MB
//   x_f16   : 67,108,864 B
//   wpack   :  1,064,960 B
//   bias    :      4,352 B (padded)
//   xproj   : 272,629,760 B
extern "C" void kernel_launch(void* const* d_in, const int* in_sizes, int n_in,
                              void* d_out, int out_size, void* d_ws, size_t ws_size,
                              hipStream_t stream) {
  const float* x_in = (const float*)d_in[0];
  const float* Wf = (const float*)d_in[1];  const float* bf = (const float*)d_in[2];
  const float* Wi = (const float*)d_in[3];  const float* bi = (const float*)d_in[4];
  const float* Wg = (const float*)d_in[5];  const float* bg = (const float*)d_in[6];
  const float* Wo = (const float*)d_in[7];  const float* bo = (const float*)d_in[8];
  const float* Wq = (const float*)d_in[9];  const float* bq = (const float*)d_in[10];

  char* ws = (char*)d_ws;
  size_t off = 0;
  _Float16* xh = (_Float16*)(ws + off);     off += (size_t)T_STEPS * BATCH * DIN * 2;
  _Float16* wpack = (_Float16*)(ws + off);  off += (size_t)2 * 8 * NT * 512 * 2;
  float* bias = (float*)(ws + off);         off += ((size_t)NPAD * 4 + 255) & ~(size_t)255;
  _Float16* xproj = (_Float16*)(ws + off);  // (size_t)MT_ALL * NT * 256 * 2 bytes

  float* out = (float*)d_out;
  float* out_hx = out + (size_t)T_STEPS * BATCH * DH;
  float* out_cx = out_hx + (size_t)BATCH * DH;

  k_convert_x<<<4096, 256, 0, stream>>>(x_in, xh, T_STEPS * BATCH * DIN);
  k_pack<<<1040, 256, 0, stream>>>(Wf, Wi, Wg, Wo, Wq, bf, bi, bg, bo, bq, wpack, bias);
  k_input_proj<<<(MT_ALL * NT) / 8, 256, 0, stream>>>(xh, wpack, bias, xproj);
  k_recurrent<<<1, 1024, 0, stream>>>(wpack + (size_t)8 * NT * 512, xproj,
                                      out, out_hx, out_cx);
}
